// MultibandFrameAttention_5566277616589
// MI455X (gfx1250) — compile-verified, matrix operand order checked by reading, not run on hardware
//
#include <hip/hip_runtime.h>
#include <hip/hip_bf16.h>

typedef __bf16 bf16_t;
typedef __bf16 v16bf __attribute__((ext_vector_type(16)));
typedef float  v8f   __attribute__((ext_vector_type(8)));

#define BINS  512
#define CROP  2048
#define BATCH 2
#define HEADS 8
#define HDIM  64
#define MROWS (BATCH * CROP)   // 4096 rows of x / o

// CDNA5 async global->LDS path (ASYNCcnt-tracked) via inline asm; the clang
// builtin's pointer address-space typing is toolchain-dependent, the ISA
// mnemonic is not. Gated on the device pass so host/other targets fall back.
#if defined(__gfx1250__)
#define ASYNC_OK 1
#define AS3 __attribute__((address_space(3)))
__device__ __forceinline__ unsigned lds_addr32(const void* p) {
  return (unsigned)(uintptr_t)(AS3 const void*)p;
}
// GLOBAL_LOAD_ASYNC_TO_LDS_B128: vdst = LDS byte address, vaddr = 64-bit VA
__device__ __forceinline__ void async_ld_b128(const void* g, void* l) {
  asm volatile("global_load_async_to_lds_b128 %0, %1, off"
               :
               : "v"(lds_addr32(l)), "v"((unsigned long long)(uintptr_t)g)
               : "memory");
}
__device__ __forceinline__ void async_wait0() {
#if __has_builtin(__builtin_amdgcn_s_wait_asynccnt)
  __builtin_amdgcn_s_wait_asynccnt(0);
#else
  asm volatile("s_wait_asynccnt 0x0" ::: "memory");
#endif
}
#else
#define ASYNC_OK 0
#endif

// ---------------------------------------------------------------------------
// D = A(16x32 bf16) x B(32x16 bf16) + C(16x16 f32), wave32 WMMA
// ---------------------------------------------------------------------------
__device__ __forceinline__ v8f wmma_bf16(v16bf a, v16bf b, v8f c) {
  return __builtin_amdgcn_wmma_f32_16x16x32_bf16(
      /*neg_a=*/false, a, /*neg_b=*/false, b,
      /*c_mod=*/(short)0, c, /*reuse_a=*/false, /*reuse_b=*/false);
}

// Load one 16x32 bf16 WMMA operand fragment from a row-major tile.
// ISA 16-bit A layout: lanes 0-15 (half=0) hold K={0..7,16..23},
// lanes 16-31 (half=1) hold K={8..15,24..31}, 2 halves per VGPR.
__device__ __forceinline__ v16bf load_frag(const bf16_t* __restrict__ base,
                                           int row, int ld, int half) {
  union { v16bf v; unsigned u[8]; } r;
  const unsigned* p = (const unsigned*)(base + row * ld) + half * 4;
#pragma unroll
  for (int i = 0; i < 4; ++i) { r.u[i] = p[i]; r.u[4 + i] = p[8 + i]; }
  return r.v;
}

// ---------------------------------------------------------------------------
// fp32 -> bf16 converts
// ---------------------------------------------------------------------------
__global__ void cvt_f32_bf16(const float* __restrict__ s,
                             bf16_t* __restrict__ d, int n) {
  for (int i = blockIdx.x * blockDim.x + threadIdx.x; i < n;
       i += gridDim.x * blockDim.x)
    d[i] = (bf16_t)s[i];
}

// er: [HDIM, CROP]  ->  erT: [CROP, HDIM] bf16 (row-major, WMMA-friendly)
__global__ void cvt_er_transpose(const float* __restrict__ er,
                                 bf16_t* __restrict__ erT) {
  int i = blockIdx.x * blockDim.x + threadIdx.x;
  if (i < CROP * HDIM) {
    int c = i >> 6, dd = i & 63;
    erT[i] = (bf16_t)er[dd * CROP + c];
  }
}

// ---------------------------------------------------------------------------
// y = A[4096x512] @ W^T[512x512] + bias.   mode 0: store bf16 into [b,h,w,d]
// (QKV); mode 1: store fp32 into [b,w,c] (final output projection).
// Block: 128 threads = 4 waves; each wave computes a 16x64 strip of a 64x64
// output tile; K consumed 32 at a time through LDS.
// ---------------------------------------------------------------------------
__global__ __launch_bounds__(128)
void gemm_proj(const bf16_t* __restrict__ A, const bf16_t* __restrict__ W,
               const float* __restrict__ bias, bf16_t* __restrict__ outp,
               float* __restrict__ outf, int mode) {
  __shared__ bf16_t Al[64 * 32];
  __shared__ bf16_t Bl[64 * 32];
  const int tid  = threadIdx.x;
  const int wave = tid >> 5, ln = tid & 15, half = (tid >> 4) & 1;
  const int gm0 = blockIdx.x * 64, gn0 = blockIdx.y * 64;

  v8f acc[4] = {};
  for (int k0 = 0; k0 < BINS; k0 += 32) {
#if ASYNC_OK
    {
      // one b128 chunk per thread per half-tile; rows split 0..31 / 32..63
      int r = tid >> 2, c = (tid & 3) << 3;
      const bf16_t* ga = A + (size_t)(gm0 + r) * BINS + k0 + c;
      const bf16_t* gw = W + (size_t)(gn0 + r) * BINS + k0 + c;
      bf16_t* la = Al + r * 32 + c;
      bf16_t* lb = Bl + r * 32 + c;
      async_ld_b128(ga, la);
      async_ld_b128(gw, lb);
      async_ld_b128(ga + 32 * BINS, la + 32 * 32);
      async_ld_b128(gw + 32 * BINS, lb + 32 * 32);
    }
    async_wait0();
#else
#pragma unroll
    for (int it = 0; it < 2; ++it) {
      int ch = tid + (it << 7);
      int r = ch >> 2, c = (ch & 3) << 3;
      *(uint4*)&Al[r * 32 + c] =
          *(const uint4*)&A[(size_t)(gm0 + r) * BINS + k0 + c];
      *(uint4*)&Bl[r * 32 + c] =
          *(const uint4*)&W[(size_t)(gn0 + r) * BINS + k0 + c];
    }
#endif
    if (k0 + 32 < BINS) {
      __builtin_prefetch(&A[(size_t)gm0 * BINS + k0 + 32], 0, 1);
      __builtin_prefetch(&W[(size_t)gn0 * BINS + k0 + 32], 0, 1);
    }
    __syncthreads();

    v16bf af = load_frag(Al + (wave << 4) * 32, ln, 32, half);
    v16bf b0 = load_frag(Bl + 0 * 16 * 32, ln, 32, half);
    v16bf b1 = load_frag(Bl + 1 * 16 * 32, ln, 32, half);
    v16bf b2 = load_frag(Bl + 2 * 16 * 32, ln, 32, half);
    v16bf b3 = load_frag(Bl + 3 * 16 * 32, ln, 32, half);
    acc[0] = wmma_bf16(af, b0, acc[0]);
    acc[1] = wmma_bf16(af, b1, acc[1]);
    acc[2] = wmma_bf16(af, b2, acc[2]);
    acc[3] = wmma_bf16(af, b3, acc[3]);
    __syncthreads();
  }

  // C layout: VGPR r, lane -> (m = r + 8*half, n = lane&15)
#pragma unroll
  for (int t = 0; t < 4; ++t) {
    int n = gn0 + (t << 4) + ln;
    float bv = bias[n];
#pragma unroll
    for (int r = 0; r < 8; ++r) {
      int   gm  = gm0 + (wave << 4) + r + (half << 3);
      float val = acc[t][r] + bv;
      if (mode == 0) {  // [b,h,w,d] bf16
        int b = gm >> 11, w = gm & 2047, hh = n >> 6, dd = n & 63;
        outp[((size_t)(b * HEADS + hh) * CROP + w) * HDIM + dd] = (bf16_t)val;
      } else {          // [b,w,c] fp32
        outf[(size_t)gm * BINS + n] = val;
      }
    }
  }
}

// ---------------------------------------------------------------------------
// Fused flash attention with Music-Transformer relative bias.
//   S[i,j] = ( q_i.k_j  +  q_j.er[:,i] ) / sqrt(512)
// grid: (CROP/64, HEADS, BATCH); block 128 = 4 waves; wave owns 16 queries.
// ---------------------------------------------------------------------------
__global__ __launch_bounds__(128)
void mb_attention(const bf16_t* __restrict__ Q, const bf16_t* __restrict__ K,
                  const bf16_t* __restrict__ V, const bf16_t* __restrict__ erT,
                  bf16_t* __restrict__ O) {
  __shared__ bf16_t Kl[32 * HDIM];   // key rows, row-major [key][d]
  __shared__ bf16_t Ql[32 * HDIM];   // q rows at key indices (for bias term)
  __shared__ bf16_t Vt[HDIM * 32];   // V transposed [d][key]
  __shared__ bf16_t Pl[4][16 * 32];  // per-wave softmax tile

  const int tid  = threadIdx.x;
  const int wave = tid >> 5, ln = tid & 15, half = (tid >> 4) & 1;
  const int h = blockIdx.y, b = blockIdx.z;
  const size_t hoff = (size_t)(b * HEADS + h) * CROP * HDIM;
  const bf16_t* Qh = Q + hoff;
  const bf16_t* Kh = K + hoff;
  const bf16_t* Vh = V + hoff;
  const int q0 = blockIdx.x * 64 + wave * 16;

  // resident A fragments: this wave's 16 query rows (K-dim 64 -> two frags)
  v16bf qa0 = load_frag(Qh  + (size_t)q0 * HDIM,      ln, HDIM, half);
  v16bf qa1 = load_frag(Qh  + (size_t)q0 * HDIM + 32, ln, HDIM, half);
  v16bf ea0 = load_frag(erT + (size_t)q0 * HDIM,      ln, HDIM, half);
  v16bf ea1 = load_frag(erT + (size_t)q0 * HDIM + 32, ln, HDIM, half);

  v8f o0 = {}, o1 = {}, o2 = {}, o3 = {};
  float mstate[8], lstate[8];
#pragma unroll
  for (int r = 0; r < 8; ++r) { mstate[r] = -3.0e38f; lstate[r] = 0.f; }
  const float inv_sqrt_c = 0.044194173824159216f;  // 1/sqrt(512)

  for (int j0 = 0; j0 < CROP; j0 += 32) {
    // ---- stage K / q-at-keys tiles (contiguous 4 KB each) ----
#if ASYNC_OK
    {
      const bf16_t* gk2 = Kh + (size_t)j0 * HDIM + tid * 8;  // 16 B / thread
      const bf16_t* gq2 = Qh + (size_t)j0 * HDIM + tid * 8;
      bf16_t* lk2 = Kl + tid * 8;
      bf16_t* lq2 = Ql + tid * 8;
      async_ld_b128(gk2, lk2);
      async_ld_b128(gk2 + 1024, lk2 + 1024);  // second 2 KB of the tile
      async_ld_b128(gq2, lq2);
      async_ld_b128(gq2 + 1024, lq2 + 1024);
    }
#else
#pragma unroll
    for (int it = 0; it < 2; ++it) {
      int ch = tid + (it << 7);
      ((uint4*)Kl)[ch] = ((const uint4*)(Kh + (size_t)j0 * HDIM))[ch];
      ((uint4*)Ql)[ch] = ((const uint4*)(Qh + (size_t)j0 * HDIM))[ch];
    }
#endif
    // ---- stage V transposed (needs element shuffle; stays synchronous) ----
    {
      const uint4* gv = (const uint4*)(Vh + (size_t)j0 * HDIM);
#pragma unroll
      for (int it = 0; it < 2; ++it) {
        int ch = tid + (it << 7);
        uint4 dv = gv[ch];
        int r = ch >> 3, c0 = (ch & 7) << 3;
        const bf16_t* hp = (const bf16_t*)&dv;
#pragma unroll
        for (int i = 0; i < 8; ++i) Vt[(c0 + i) * 32 + r] = hp[i];
      }
    }
    if (j0 + 32 < CROP) {
      __builtin_prefetch(Kh + (size_t)(j0 + 32) * HDIM, 0, 1);
      __builtin_prefetch(Vh + (size_t)(j0 + 32) * HDIM, 0, 1);
    }
#if ASYNC_OK
    async_wait0();
#endif
    __syncthreads();

    // ---- scores: S = Q.K^T + ErT.Q_keys^T ----
    v8f s0 = {}, s1 = {};
    {
      v16bf t0 = load_frag(Kl,                  ln, HDIM, half);
      v16bf t1 = load_frag(Kl + 32,             ln, HDIM, half);
      v16bf t2 = load_frag(Kl + 16 * HDIM,      ln, HDIM, half);
      v16bf t3 = load_frag(Kl + 16 * HDIM + 32, ln, HDIM, half);
      s0 = wmma_bf16(qa0, t0, s0);  s0 = wmma_bf16(qa1, t1, s0);
      s1 = wmma_bf16(qa0, t2, s1);  s1 = wmma_bf16(qa1, t3, s1);
      t0 = load_frag(Ql,                  ln, HDIM, half);
      t1 = load_frag(Ql + 32,             ln, HDIM, half);
      t2 = load_frag(Ql + 16 * HDIM,      ln, HDIM, half);
      t3 = load_frag(Ql + 16 * HDIM + 32, ln, HDIM, half);
      s0 = wmma_bf16(ea0, t0, s0);  s0 = wmma_bf16(ea1, t1, s0);
      s1 = wmma_bf16(ea0, t2, s1);  s1 = wmma_bf16(ea1, t3, s1);
    }
#pragma unroll
    for (int r = 0; r < 8; ++r) { s0[r] *= inv_sqrt_c; s1[r] *= inv_sqrt_c; }

    // ---- online softmax over this 32-key chunk ----
    bf16_t* Pw = Pl[wave];
#pragma unroll
    for (int r = 0; r < 8; ++r) {
      float t = fmaxf(s0[r], s1[r]);
      t = fmaxf(t, __shfl_xor(t, 1, 32));
      t = fmaxf(t, __shfl_xor(t, 2, 32));
      t = fmaxf(t, __shfl_xor(t, 4, 32));
      t = fmaxf(t, __shfl_xor(t, 8, 32));
      float mnew  = fmaxf(mstate[r], t);
      float alpha = __expf(mstate[r] - mnew);
      float p0 = __expf(s0[r] - mnew);
      float p1 = __expf(s1[r] - mnew);
      float rs = p0 + p1;
      rs += __shfl_xor(rs, 1, 32);
      rs += __shfl_xor(rs, 2, 32);
      rs += __shfl_xor(rs, 4, 32);
      rs += __shfl_xor(rs, 8, 32);
      lstate[r] = lstate[r] * alpha + rs;
      mstate[r] = mnew;
      o0[r] *= alpha; o1[r] *= alpha; o2[r] *= alpha; o3[r] *= alpha;
      int m = r + (half << 3);
      Pw[m * 32 + ln]      = (bf16_t)p0;
      Pw[m * 32 + 16 + ln] = (bf16_t)p1;
    }
    __syncthreads();  // P visible as A-fragment; all waves done with Kl/Ql

    // ---- O += P x V ----
    v16bf pf = load_frag(Pw, ln, 32, half);
    v16bf v0 = load_frag(Vt,           ln, 32, half);
    v16bf v1 = load_frag(Vt + 16 * 32, ln, 32, half);
    v16bf v2 = load_frag(Vt + 32 * 32, ln, 32, half);
    v16bf v3 = load_frag(Vt + 48 * 32, ln, 32, half);
    o0 = wmma_bf16(pf, v0, o0);
    o1 = wmma_bf16(pf, v1, o1);
    o2 = wmma_bf16(pf, v2, o2);
    o3 = wmma_bf16(pf, v3, o3);
    __syncthreads();  // protect LDS tiles before next chunk's staging
  }

  // ---- epilogue: normalize, store bf16 to [b, w, c] for output projection
#pragma unroll
  for (int r = 0; r < 8; ++r) {
    float invl = 1.0f / lstate[r];
    int m = r + (half << 3);
    size_t rowbase = ((size_t)b * CROP + (q0 + m)) * BINS + h * HDIM;
    O[rowbase +  0 + ln] = (bf16_t)(o0[r] * invl);
    O[rowbase + 16 + ln] = (bf16_t)(o1[r] * invl);
    O[rowbase + 32 + ln] = (bf16_t)(o2[r] * invl);
    O[rowbase + 48 + ln] = (bf16_t)(o3[r] * invl);
  }
}

// ---------------------------------------------------------------------------
extern "C" void kernel_launch(void* const* d_in, const int* in_sizes, int n_in,
                              void* d_out, int out_size, void* d_ws,
                              size_t ws_size, hipStream_t stream) {
  (void)in_sizes; (void)n_in; (void)out_size; (void)ws_size;
  const float* x  = (const float*)d_in[0];
  const float* Wq = (const float*)d_in[1];
  const float* bq = (const float*)d_in[2];
  const float* Wk = (const float*)d_in[3];
  const float* bk = (const float*)d_in[4];
  const float* Wv = (const float*)d_in[5];
  const float* bv = (const float*)d_in[6];
  const float* Wo = (const float*)d_in[7];
  const float* bo = (const float*)d_in[8];
  const float* er = (const float*)d_in[9];
  float* out = (float*)d_out;

  char* ws = (char*)d_ws;
  size_t off = 0;
  auto alloc = [&](size_t bytes) {
    char* p = ws + off;
    off += (bytes + 255) & ~(size_t)255;
    return p;
  };
  bf16_t* x_bf  = (bf16_t*)alloc((size_t)MROWS * BINS * 2);  //  4 MB
  bf16_t* wq_bf = (bf16_t*)alloc((size_t)BINS * BINS * 2);   // 512 KB
  bf16_t* wk_bf = (bf16_t*)alloc((size_t)BINS * BINS * 2);
  bf16_t* wv_bf = (bf16_t*)alloc((size_t)BINS * BINS * 2);
  bf16_t* wo_bf = (bf16_t*)alloc((size_t)BINS * BINS * 2);
  bf16_t* erT_bf = (bf16_t*)alloc((size_t)CROP * HDIM * 2);  // 256 KB
  bf16_t* q_bf  = (bf16_t*)alloc((size_t)BATCH * HEADS * CROP * HDIM * 2); // 4 MB
  bf16_t* k_bf  = (bf16_t*)alloc((size_t)BATCH * HEADS * CROP * HDIM * 2);
  bf16_t* v_bf  = (bf16_t*)alloc((size_t)BATCH * HEADS * CROP * HDIM * 2);
  bf16_t* o_bf  = (bf16_t*)alloc((size_t)MROWS * BINS * 2);  //  4 MB

  int n;
  n = MROWS * BINS;
  cvt_f32_bf16<<<(n + 255) / 256, 256, 0, stream>>>(x, x_bf, n);
  n = BINS * BINS;
  cvt_f32_bf16<<<(n + 255) / 256, 256, 0, stream>>>(Wq, wq_bf, n);
  cvt_f32_bf16<<<(n + 255) / 256, 256, 0, stream>>>(Wk, wk_bf, n);
  cvt_f32_bf16<<<(n + 255) / 256, 256, 0, stream>>>(Wv, wv_bf, n);
  cvt_f32_bf16<<<(n + 255) / 256, 256, 0, stream>>>(Wo, wo_bf, n);
  n = CROP * HDIM;
  cvt_er_transpose<<<(n + 255) / 256, 256, 0, stream>>>(er, erT_bf);

  dim3 gg(MROWS / 64, BINS / 64);
  gemm_proj<<<gg, 128, 0, stream>>>(x_bf, wq_bf, bq, q_bf, nullptr, 0);
  gemm_proj<<<gg, 128, 0, stream>>>(x_bf, wk_bf, bk, k_bf, nullptr, 0);
  gemm_proj<<<gg, 128, 0, stream>>>(x_bf, wv_bf, bv, v_bf, nullptr, 0);

  dim3 ga(CROP / 64, HEADS, BATCH);
  mb_attention<<<ga, 128, 0, stream>>>(q_bf, k_bf, v_bf, erT_bf, o_bf);

  gemm_proj<<<gg, 128, 0, stream>>>(o_bf, wo_bf, bo, nullptr, out, 1);
}
